// TriangleMultiplicativeModuleNew_76020921139697
// MI455X (gfx1250) — matrix operand order repaired and checked by reading, not compiled, not run on hardware
//
#include <hip/hip_runtime.h>
#include <hip/hip_bf16.h>

typedef __attribute__((ext_vector_type(16))) _Float16 v16h;
typedef __attribute__((ext_vector_type(8)))  _Float16 h8;
typedef __attribute__((ext_vector_type(8)))  float    v8f;

#define NSEQ   384
#define DMODEL 128
#define NROW   (NSEQ * NSEQ)   /* 147456 */

__device__ __forceinline__ float sigmoidf_(float v) { return 1.0f / (1.0f + __expf(-v)); }

__device__ __forceinline__ v16h pack16(h8 lo, h8 hi) {
    v16h a;
#pragma unroll
    for (int t = 0; t < 8; ++t) { a[t] = lo[t]; a[t + 8] = hi[t]; }
    return a;
}

// Pre-swizzled B-fragment pointer: [mat][nt(8)][kt(4)][lane(32)][16 halfs]
__device__ __forceinline__ const _Float16* wfrag(const _Float16* wf, int mat, int nt, int kt, int lane) {
    return wf + ((size_t)(((mat * 8 + nt) * 4 + kt) * 32 + lane) * 16);
}

// ---------------------------------------------------------------------------
// Kernel 0: swizzle 6 weight matrices (f32 [128][128], K-major rows) into
// WMMA B-fragment layout in f16. Grid: 192 blocks (mat*32 + nt*4 + kt) x 32.
// B frag (16x16x32 f16): lanes 0-15 hold K=0..7,16..23 of column n; lanes 16-31 K=8..15,24..31.
// ---------------------------------------------------------------------------
__global__ void k_prep(const float* __restrict__ Wl, const float* __restrict__ Wr,
                       const float* __restrict__ Wlg, const float* __restrict__ Wrg,
                       const float* __restrict__ Wog, const float* __restrict__ Wo,
                       _Float16* __restrict__ wf) {
    int frag = blockIdx.x;            // 0..191
    int mat  = frag >> 5;             // 0..5
    int idx  = frag & 31;
    int nt = idx >> 2, kt = idx & 3;
    const float* W = (mat == 0) ? Wl : (mat == 1) ? Wr : (mat == 2) ? Wlg
                   : (mat == 3) ? Wrg : (mat == 4) ? Wog : Wo;
    int lane  = threadIdx.x;
    int n     = nt * 16 + (lane & 15);
    int kbase = kt * 32 + (lane >> 4) * 8;
    v16h val;
#pragma unroll
    for (int t = 0; t < 16; ++t) val[t] = (_Float16)W[(size_t)(kbase + (t < 8 ? t : t + 8)) * DMODEL + n];
    *(v16h*)(wf + ((size_t)frag * 32 + lane) * 16) = val;
}

// ---------------------------------------------------------------------------
// Kernel 1: fused LayerNorm + 5 projections + gating.
//   left_t[d][r]  = (xn@Wl+bl)*sigmoid(xn@Wlg+blg)   (f16, channel-major)
//   right_t[d][r] = (xn@Wr+br)*sigmoid(xn@Wrg+brg)   (f16, channel-major)
//   gate[r][d]    = sigmoid(xn@Wog+bog)              (f32, row-major)
// Block = 256 threads = 8 waves; each wave owns one 16-row tile.
// ---------------------------------------------------------------------------
__global__ __launch_bounds__(256) void k_proj(
    const float* __restrict__ x, const float* __restrict__ na, const float* __restrict__ nb,
    const _Float16* __restrict__ wf,
    const float* __restrict__ bl, const float* __restrict__ br,
    const float* __restrict__ blg, const float* __restrict__ brg, const float* __restrict__ bog,
    _Float16* __restrict__ left_t, _Float16* __restrict__ right_t, float* __restrict__ gate) {
    __shared__ _Float16 xn[8][16][136];   // f16 LN'ed tile per wave (+pad, rows 272B aligned)
    __shared__ _Float16 tsc[8][16][24];   // transpose scratch per wave (rows 48B aligned)

    int tid = threadIdx.x;
    int wave = tid >> 5, lane = tid & 31;
    int r0 = (blockIdx.x * 8 + wave) * 16;

    // ---- LayerNorm over 128 (two lanes per row, 64 elems each) ----
    int row = lane >> 1;
    int hsel = lane & 1;
    const float* xp = x + (size_t)(r0 + row) * DMODEL + hsel * 64;
    float s = 0.f, ss = 0.f;
#pragma unroll
    for (int t = 0; t < 16; ++t) {
        float4 v = ((const float4*)xp)[t];
        s  += v.x + v.y + v.z + v.w;
        ss += v.x * v.x + v.y * v.y + v.z * v.z + v.w * v.w;
    }
    s  += __shfl_xor(s, 1);
    ss += __shfl_xor(ss, 1);
    float mean = s * (1.0f / 128.0f);
    float var  = ss * (1.0f / 128.0f) - mean * mean;
    float rinv = rsqrtf(var + 1e-5f);
#pragma unroll
    for (int t = 0; t < 16; ++t) {
        float4 v = ((const float4*)xp)[t];
        int c = hsel * 64 + t * 4;
        xn[wave][row][c + 0] = (_Float16)((v.x - mean) * rinv * na[c + 0] + nb[c + 0]);
        xn[wave][row][c + 1] = (_Float16)((v.y - mean) * rinv * na[c + 1] + nb[c + 1]);
        xn[wave][row][c + 2] = (_Float16)((v.z - mean) * rinv * na[c + 2] + nb[c + 2]);
        xn[wave][row][c + 3] = (_Float16)((v.w - mean) * rinv * na[c + 3] + nb[c + 3]);
    }
    __syncthreads();

    // ---- A fragments (ISA 16-bit A 16x32 layout) ----
    int m = lane & 15, hs = lane >> 4;
    v16h afr[4];
#pragma unroll
    for (int kt = 0; kt < 4; ++kt) {
        const _Float16* p = &xn[wave][m][kt * 32 + hs * 8];
        afr[kt] = pack16(*(const h8*)p, *(const h8*)(p + 16));
    }
    int n = lane & 15;
    int mrow_off = hs * 8;   // C/D layout: lanes 16-31 hold rows M+8

    // ---- left / right: P * sigmoid(G), transposed store to [d][r] ----
#pragma unroll
    for (int pair = 0; pair < 2; ++pair) {
        const float* bP = pair ? br : bl;
        const float* bG = pair ? brg : blg;
        _Float16* dst = pair ? right_t : left_t;
        int mp = pair, mg = 2 + pair;
        for (int nt = 0; nt < 8; ++nt) {
            v8f ap = {}, ag = {};
#pragma unroll
            for (int kt = 0; kt < 4; ++kt) {
                v16h bfp = *(const v16h*)wfrag(wf, mp, nt, kt, lane);
                v16h bfg = *(const v16h*)wfrag(wf, mg, nt, kt, lane);
                ap = __builtin_amdgcn_wmma_f32_16x16x32_f16(false, afr[kt], false, bfp, (short)0, ap, false, false);
                ag = __builtin_amdgcn_wmma_f32_16x16x32_f16(false, afr[kt], false, bfg, (short)0, ag, false, false);
            }
            float biasP = bP[nt * 16 + n];
            float biasG = bG[nt * 16 + n];
#pragma unroll
            for (int v = 0; v < 8; ++v) {
                float ov = (ap[v] + biasP) * sigmoidf_(ag[v] + biasG);
                tsc[wave][n][v + mrow_off] = (_Float16)ov;   // [d-in-tile][row]
            }
            __syncthreads();
            {   // lane<16 writes halves 0..7, lane>=16 halves 8..15 of a 16-run along r
                int d = nt * 16 + n;
                const _Float16* sp = &tsc[wave][n][mrow_off];
                h8 val = *(const h8*)sp;
                *(h8*)(dst + (size_t)d * NROW + r0 + mrow_off) = val;
            }
            __syncthreads();
        }
    }

    // ---- out_gate: sigmoid, natural [r][d] f32 ----
    for (int nt = 0; nt < 8; ++nt) {
        v8f ag = {};
#pragma unroll
        for (int kt = 0; kt < 4; ++kt) {
            v16h bfg = *(const v16h*)wfrag(wf, 4, nt, kt, lane);
            ag = __builtin_amdgcn_wmma_f32_16x16x32_f16(false, afr[kt], false, bfg, (short)0, ag, false, false);
        }
        float biasG = bog[nt * 16 + n];
#pragma unroll
        for (int v = 0; v < 8; ++v) {
            int rr = v + mrow_off;
            gate[(size_t)(r0 + rr) * DMODEL + nt * 16 + n] = sigmoidf_(ag[v] + biasG);
        }
    }
}

// ---------------------------------------------------------------------------
// Kernel 2: batched per-channel GEMM  OUT_d = (R_d^T @ L_d) / n
//   R_d = right_t[d] (384x384, row=k, col=i), L_d = left_t[d] (row=k, col=j)
// Grid: (3, 6, 128); block tile 128(i) x 64(j); 8 waves in 4x2, 32x32 each.
// K-panels are DMA'd with double-buffered global_load_async_to_lds_b128
// (ASYNCcnt pipeline), then transposed LDS->LDS into fragment-friendly tiles.
// ---------------------------------------------------------------------------
__global__ __launch_bounds__(256) void k_einsum(
    const _Float16* __restrict__ left_t, const _Float16* __restrict__ right_t,
    float* __restrict__ out_t) {
    __shared__ _Float16 praw[2][32][192];  // [buf][k][0..127: R cols | 128..191: L cols] rows 384B
    __shared__ _Float16 Rt[128][40];       // [i][kk], rows 80B (16B aligned)
    __shared__ _Float16 Lt[64][40];        // [j][kk]

    int tid = threadIdx.x, wave = tid >> 5, lane = tid & 31;
    int d  = blockIdx.z;
    int i0 = blockIdx.x * 128, j0 = blockIdx.y * 64;
    const _Float16* Rch = right_t + (size_t)d * NROW;
    const _Float16* Lch = left_t  + (size_t)d * NROW;
    int wi = wave & 3, wj = wave >> 2;
    v8f acc[2][2] = {};
    int m = lane & 15, hs = lane >> 4, n = lane & 15;

    // issue one panel's async DMA batch (3 x b128 per thread = 3 ASYNCcnt per wave)
    auto stage = [&](int buf, int k0) {
#pragma unroll
        for (int q = 0; q < 3; ++q) {
            int chunk = q * 256 + tid;      // 0..767 16-byte chunks
            int row  = chunk / 24;          // k within panel (0..31)
            int slot = chunk % 24;          // 16B slot within 384B row
            const _Float16* g = (slot < 16)
                ? (Rch + (size_t)(k0 + row) * NSEQ + i0 + slot * 8)
                : (Lch + (size_t)(k0 + row) * NSEQ + j0 + (slot - 16) * 8);
            unsigned ldsa = (unsigned)(uintptr_t)&praw[buf][row][slot * 8];
            unsigned long long ga = (unsigned long long)(uintptr_t)g;
            asm volatile("global_load_async_to_lds_b128 %0, %1, off"
                         :: "v"(ldsa), "v"(ga) : "memory");
        }
    };

    stage(0, 0);
    for (int s = 0; s < NSEQ / 32; ++s) {
        int buf = s & 1;
        if (s + 1 < NSEQ / 32) {
            stage(buf ^ 1, (s + 1) * 32);                       // prefetch next panel
            asm volatile("s_wait_asynccnt 0x3" ::: "memory");   // current panel landed
        } else {
            asm volatile("s_wait_asynccnt 0x0" ::: "memory");
        }
        __syncthreads();   // all waves' async stores to praw[buf] visible

        {   // transpose praw[buf] -> Rt / Lt (contiguous LDS reads, scattered writes)
            int k = tid >> 3, g = tid & 7;
            const _Float16* src = &praw[buf][k][g * 24];
#pragma unroll
            for (int c = 0; c < 24; ++c) {
                int col = g * 24 + c;
                _Float16 v = src[c];
                if (col < 128) Rt[col][k] = v;
                else           Lt[col - 128][k] = v;
            }
        }
        __syncthreads();

        v16h af[2], bf[2];
#pragma unroll
        for (int ti = 0; ti < 2; ++ti) {
            const _Float16* p = &Rt[wi * 32 + ti * 16 + m][hs * 8];
            af[ti] = pack16(*(const h8*)p, *(const h8*)(p + 16));
        }
#pragma unroll
        for (int tj = 0; tj < 2; ++tj) {
            const _Float16* p = &Lt[wj * 32 + tj * 16 + n][hs * 8];
            bf[tj] = pack16(*(const h8*)p, *(const h8*)(p + 16));
        }
        __syncthreads();   // frags in regs; Rt/Lt free for next step

#pragma unroll
        for (int ti = 0; ti < 2; ++ti)
#pragma unroll
            for (int tj = 0; tj < 2; ++tj)
                acc[ti][tj] = __builtin_amdgcn_wmma_f32_16x16x32_f16(
                    false, af[ti], false, bf[tj], (short)0, acc[ti][tj], false, false);
    }

    const float scale = 1.0f / (float)NSEQ;
    int ms = hs * 8;
#pragma unroll
    for (int ti = 0; ti < 2; ++ti)
#pragma unroll
        for (int tj = 0; tj < 2; ++tj)
#pragma unroll
            for (int v = 0; v < 8; ++v) {
                int i = i0 + wi * 32 + ti * 16 + v + ms;
                int j = j0 + wj * 32 + tj * 16 + n;
                out_t[(size_t)d * NROW + (size_t)i * NSEQ + j] = acc[ti][tj][v] * scale;
            }
}

// ---------------------------------------------------------------------------
// Kernel 3: LN over d + out_gate + final GEMM (@Wo) + bias -> fp32 output.
// Block handles 64 rows; out_t staged transposed in LDS, f16 A tile rebuilt.
// ---------------------------------------------------------------------------
__global__ __launch_bounds__(256) void k_out(
    const float* __restrict__ out_t, const float* __restrict__ gate,
    const float* __restrict__ ona, const float* __restrict__ onb,
    const _Float16* __restrict__ wf, const float* __restrict__ bo,
    float* __restrict__ out) {
    __shared__ float    X[64][132];    // [r][d] fp32
    __shared__ _Float16 Af[64][136];   // [r][d] f16 normalized*gated

    int tid = threadIdx.x, wave = tid >> 5, lane = tid & 31;
    int r0 = blockIdx.x * 64;

    {   // stage transpose: thread -> (d = tid/2, 32-run of r)
        int d = tid >> 1, rb = (tid & 1) * 32;
        const float* src = out_t + (size_t)d * NROW + r0 + rb;
#pragma unroll
        for (int t = 0; t < 32; t += 4) {
            float4 v = *(const float4*)(src + t);
            X[rb + t + 0][d] = v.x; X[rb + t + 1][d] = v.y;
            X[rb + t + 2][d] = v.z; X[rb + t + 3][d] = v.w;
        }
    }
    __syncthreads();

    {   // LN over d: 4 lanes per row, 32 elems each
        int r = tid >> 2, q = tid & 3;
        float s = 0.f, ss = 0.f;
#pragma unroll
        for (int t = 0; t < 32; ++t) { float v = X[r][q * 32 + t]; s += v; ss += v * v; }
        s  += __shfl_xor(s, 1);  ss += __shfl_xor(ss, 1);
        s  += __shfl_xor(s, 2);  ss += __shfl_xor(ss, 2);
        float mean = s * (1.0f / 128.0f);
        float var  = ss * (1.0f / 128.0f) - mean * mean;
        float rinv = rsqrtf(var + 1e-5f);
        const float* gp = gate + (size_t)(r0 + r) * DMODEL + q * 32;
#pragma unroll
        for (int t = 0; t < 32; ++t) {
            int c = q * 32 + t;
            float v = (X[r][c] - mean) * rinv * ona[c] + onb[c];
            Af[r][c] = (_Float16)(v * gp[t]);
        }
    }
    __syncthreads();

    // GEMM: wave -> (16-row mtile, half of the 8 n-tiles)
    int mt = wave >> 1, nh = wave & 1;
    int m = lane & 15, hs = lane >> 4;
    v16h afr[4];
#pragma unroll
    for (int kt = 0; kt < 4; ++kt) {
        const _Float16* p = &Af[mt * 16 + m][kt * 32 + hs * 8];
        afr[kt] = pack16(*(const h8*)p, *(const h8*)(p + 16));
    }
    int n = lane & 15, ms = hs * 8;
#pragma unroll
    for (int nt2 = 0; nt2 < 4; ++nt2) {
        int nt = nh * 4 + nt2;
        v8f acc = {};
#pragma unroll
        for (int kt = 0; kt < 4; ++kt) {
            v16h bf = *(const v16h*)wfrag(wf, 5, nt, kt, lane);
            acc = __builtin_amdgcn_wmma_f32_16x16x32_f16(false, afr[kt], false, bf, (short)0, acc, false, false);
        }
        int c = nt * 16 + n;
        float bias = bo[c];
#pragma unroll
        for (int v = 0; v < 8; ++v)
            out[(size_t)(r0 + mt * 16 + v + ms) * DMODEL + c] = acc[v] + bias;
    }
}

// ---------------------------------------------------------------------------
extern "C" void kernel_launch(void* const* d_in, const int* in_sizes, int n_in,
                              void* d_out, int out_size, void* d_ws, size_t ws_size,
                              hipStream_t stream) {
    (void)in_sizes; (void)n_in; (void)out_size; (void)ws_size;
    const float* x    = (const float*)d_in[0];
    const float* na   = (const float*)d_in[1];
    const float* nb   = (const float*)d_in[2];
    const float* Wl   = (const float*)d_in[3];
    const float* bl   = (const float*)d_in[4];
    const float* Wr   = (const float*)d_in[5];
    const float* br   = (const float*)d_in[6];
    const float* Wlg  = (const float*)d_in[7];
    const float* blg  = (const float*)d_in[8];
    const float* Wrg  = (const float*)d_in[9];
    const float* brg  = (const float*)d_in[10];
    const float* Wog  = (const float*)d_in[11];
    const float* bog  = (const float*)d_in[12];
    const float* ona  = (const float*)d_in[13];
    const float* onb  = (const float*)d_in[14];
    const float* Wo   = (const float*)d_in[15];
    const float* bo   = (const float*)d_in[16];
    float* out = (float*)d_out;

    char* ws = (char*)d_ws;
    _Float16* left_t  = (_Float16*)(ws);                       // 37,748,736 B
    _Float16* right_t = (_Float16*)(ws + 37748736ull);         // 37,748,736 B
    float*    out_t   = (float*)   (ws + 75497472ull);         // 75,497,472 B
    float*    gate    = (float*)   (ws + 150994944ull);        // 75,497,472 B
    _Float16* wf      = (_Float16*)(ws + 226492416ull);        //    196,608 B

    k_prep  <<<dim3(192),          dim3(32),  0, stream>>>(Wl, Wr, Wlg, Wrg, Wog, Wo, wf);
    k_proj  <<<dim3(NROW / 128),   dim3(256), 0, stream>>>(x, na, nb, wf, bl, br, blg, brg, bog,
                                                           left_t, right_t, gate);
    k_einsum<<<dim3(3, 6, DMODEL), dim3(256), 0, stream>>>(left_t, right_t, out_t);
    k_out   <<<dim3(NROW / 64),    dim3(256), 0, stream>>>(out_t, gate, ona, onb, wf, bo, out);
}
